// GNN_v6_5927054868949
// MI455X (gfx1250) — compile-verified, matrix operand order checked by reading, not run on hardware
//
#include <hip/hip_runtime.h>
#include <hip/hip_bf16.h>
#include <math.h>

// ---------------------------------------------------------------------------
// PointNetConv x2 + pooling for MI455X (gfx1250, wave32, WMMA).
// Edge/node MLPs run as 16-row WMMA tiles: v_wmma_f32_16x16x32_f16,
// f16 operands / f32 accumulation. Scatter-max via global_atomic_max_u32 on a
// monotone uint encoding of float (deterministic: max is associative).
// ELU uses the hardware transcendental (v_exp_f32) branchlessly.
//
// Input flattening assumption (setup_inputs() insertion order, params dict in
// insertion order local1,global1,local2,global2,lin1; each Linear = (W,b)):
//   0:x 1:pos 2..25: 12x(W,b) 26:lin1.W 27:lin1.b 28:edge_index 29:batch
// ---------------------------------------------------------------------------

typedef __attribute__((ext_vector_type(16))) _Float16 v16h;
typedef __attribute__((ext_vector_type(8)))  _Float16 v8h;
typedef __attribute__((ext_vector_type(8)))  float    v8f;

#define DEVI static __device__ __forceinline__

DEVI unsigned encf(float f) {           // order-preserving float->uint
  unsigned u = __float_as_uint(f);
  return (u & 0x80000000u) ? ~u : (u | 0x80000000u);
}
DEVI float decf(unsigned u) {
  return (u & 0x80000000u) ? __uint_as_float(u ^ 0x80000000u)
                           : __uint_as_float(~u);
}
// Branchless ELU: v_exp_f32 (TRANS32, co-executes with VALU) + v_cndmask.
DEVI float eluf(float v) {
  float em = __expf(v) - 1.0f;
  return v > 0.f ? v : em;
}

DEVI v8f wmma16(v16h a, v16h b, v8f c) {
  // D = A(16x32 f16) * B(32x16 f16) + C(16x16 f32)
  return __builtin_amdgcn_wmma_f32_16x16x32_f16(false, a, false, b,
                                                (short)0, c, false, false);
}

// A-fragment: lane holds row M=lane%16; K-halves 0-7/16-23 (lanes 0-15) or
// 8-15/24-31 (lanes 16-31) -> two contiguous 8xf16 (16B) LDS loads.
DEVI v16h load_a_frag(const _Float16* row, int half) {
  v8h lo = *(const v8h*)(row + 8 * half);
  v8h hi = *(const v8h*)(row + 16 + 8 * half);
  return __builtin_shufflevector(lo, hi, 0, 1, 2, 3, 4, 5, 6, 7,
                                 8, 9, 10, 11, 12, 13, 14, 15);
}
// B-fragment: pre-packed so each lane's 16 f16 are contiguous (32B load).
DEVI v16h load_b_frag(const _Float16* __restrict__ w16, int frag, int lane) {
  return *(const v16h*)(w16 + (size_t)frag * 512 + lane * 16);
}

#define ENC_NEG_INF 0x007FFFFFu

// ------------------------------- fill --------------------------------------
__global__ void gnn_fill(unsigned* __restrict__ p, unsigned v, int n) {
  int i = blockIdx.x * blockDim.x + threadIdx.x;
  if (i < n) p[i] = v;
}

// --------------------------- weight packing --------------------------------
// 12 matrices -> 48 B-fragments of 512 f16 each, in WMMA B-layout:
// frag(K-tile,N-tile); lane L, elem j: k = (v<4 ? 2v : 16+2(v-4)) + 8*(L/16)
// + (j&1) + 32*ktile, n = 16*ntile + L%16; zero-pad beyond (Kact,Nact).
struct PackArgs { const float* w[12]; };

__global__ void gnn_pack(_Float16* __restrict__ w16, PackArgs pa) {
  int t = blockIdx.x * blockDim.x + threadIdx.x;
  if (t >= 48 * 512) return;
  int f = t >> 9, r = t & 511, L = r >> 4, j = r & 15;
  int mat, ktile, ntile;
  if (f < 12) { mat = f >> 1; ktile = 0; ntile = f & 1; }         // 32x32 mats
  else { int f2 = f - 12; mat = 6 + f2 / 6; int wi = f2 % 6;       // 64x48 mats
         ktile = wi / 3; ntile = wi % 3; }
  int Kact = (mat == 0) ? 6 : (mat < 6 ? 32 : 35);
  int Nact = (mat < 6) ? 32 : 35;
  int half = L >> 4, v = j >> 1, p = j & 1;
  int kk = ((v < 4) ? 2 * v : 16 + 2 * (v - 4)) + 8 * half + p;
  int k = ktile * 32 + kk;
  int n = ntile * 16 + (L & 15);
  float val = (k < Kact && n < Nact) ? pa.w[mat][k * Nact + n] : 0.f;
  w16[t] = (_Float16)val;
}

// --------------------------- edge conv, layer 1 ----------------------------
// msg = mlp3_6_32(concat(x[src], pos[src]-pos[dst])); agg1[dst] = max(msg)
__global__ void __launch_bounds__(256)
gnn_edge1(const float* __restrict__ x, const float* __restrict__ pos,
          const int* __restrict__ ei, const _Float16* __restrict__ w16,
          const float* __restrict__ b0, const float* __restrict__ b1,
          const float* __restrict__ b2, unsigned* __restrict__ agg1,
          int E, int Nn) {
  __shared__ _Float16 act[8][16][32];
  __shared__ int meta[8][16];
  const int lane = threadIdx.x & 31, wv = threadIdx.x >> 5;
  const int col = lane & 15, half = lane >> 4;
  const int Etot = E + Nn;
  const int nG = (Etot + 15) >> 4;
  const int wstride = (gridDim.x * blockDim.x) >> 5;

  float bias[3][2];
  bias[0][0] = b0[col]; bias[0][1] = b0[16 + col];
  bias[1][0] = b1[col]; bias[1][1] = b1[16 + col];
  bias[2][0] = b2[col]; bias[2][1] = b2[16 + col];

  v16h Bf[6];                                   // mats 0..2 -> frags 0..5
#pragma unroll
  for (int i = 0; i < 6; ++i) Bf[i] = load_b_frag(w16, i, lane);

  for (int g = blockIdx.x * (blockDim.x >> 5) + wv; g < nG; g += wstride) {
    int e = g * 16 + col;
    if (lane < 16) {                            // stage 6 features -> LDS row
      bool valid = e < Etot;
      int src = 0, dst = 0;
      if (e < E) { src = ei[e]; dst = ei[E + e]; }
      else if (valid) { src = dst = e - E; }    // self-loop
      meta[wv][col] = valid ? dst : -1;
      float f0 = 0, f1 = 0, f2 = 0, f3 = 0, f4 = 0, f5 = 0;
      if (valid) {
        f0 = x[src * 3 + 0]; f1 = x[src * 3 + 1]; f2 = x[src * 3 + 2];
        f3 = pos[src * 3 + 0] - pos[dst * 3 + 0];
        f4 = pos[src * 3 + 1] - pos[dst * 3 + 1];
        f5 = pos[src * 3 + 2] - pos[dst * 3 + 2];
      }
      _Float16* row = &act[wv][col][0];
      row[0] = (_Float16)f0; row[1] = (_Float16)f1; row[2] = (_Float16)f2;
      row[3] = (_Float16)f3; row[4] = (_Float16)f4; row[5] = (_Float16)f5;
      row[6] = (_Float16)0.f; row[7] = (_Float16)0.f;
      int pe = e + (wstride << 4);              // prefetch next group's edges
      if (pe < E) { __builtin_prefetch(ei + pe, 0, 1);
                    __builtin_prefetch(ei + E + pe, 0, 1); }
    } else {                                    // zero K pad cols 8..31
      v8h z = {0, 0, 0, 0, 0, 0, 0, 0};
      _Float16* row = &act[wv][col][0];
      *(v8h*)(row + 8) = z; *(v8h*)(row + 16) = z; *(v8h*)(row + 24) = z;
    }
    __builtin_amdgcn_wave_barrier();

    v16h a = load_a_frag(&act[wv][col][0], half);
    v8f zc = {0.f, 0.f, 0.f, 0.f, 0.f, 0.f, 0.f, 0.f};
    v8f c0 = zc, c1 = zc;
#pragma unroll
    for (int l = 0; l < 3; ++l) {
      c0 = wmma16(a, Bf[2 * l + 0], zc);
      c1 = wmma16(a, Bf[2 * l + 1], zc);
#pragma unroll
      for (int i = 0; i < 8; ++i) { c0[i] += bias[l][0]; c1[i] += bias[l][1]; }
      if (l < 2) {                              // ELU + C-layout -> LDS -> A
#pragma unroll
        for (int i = 0; i < 8; ++i) {
          int rr = i + 8 * half;
          act[wv][rr][col]      = (_Float16)eluf(c0[i]);
          act[wv][rr][16 + col] = (_Float16)eluf(c1[i]);
        }
        __builtin_amdgcn_wave_barrier();
        a = load_a_frag(&act[wv][col][0], half);
      }
    }
#pragma unroll
    for (int i = 0; i < 8; ++i) {               // scatter-max to agg1[dst]
      int d = meta[wv][i + 8 * half];
      if (d >= 0) {
        atomicMax(&agg1[(size_t)d * 32 + col],      encf(c0[i]));
        atomicMax(&agg1[(size_t)d * 32 + 16 + col], encf(c1[i]));
      }
    }
  }
}

// ------------------------ node global MLP, layer 1 -------------------------
// h1 = f16( elu( mlp3_32_32( dec(agg1) ) ) )
__global__ void __launch_bounds__(256)
gnn_node1(const unsigned* __restrict__ agg1, const _Float16* __restrict__ w16,
          const float* __restrict__ b0, const float* __restrict__ b1,
          const float* __restrict__ b2, _Float16* __restrict__ h1, int Nn) {
  __shared__ _Float16 act[8][16][32];
  const int lane = threadIdx.x & 31, wv = threadIdx.x >> 5;
  const int col = lane & 15, half = lane >> 4;
  const int nG = (Nn + 15) >> 4;
  const int wstride = (gridDim.x * blockDim.x) >> 5;

  float bias[3][2];
  bias[0][0] = b0[col]; bias[0][1] = b0[16 + col];
  bias[1][0] = b1[col]; bias[1][1] = b1[16 + col];
  bias[2][0] = b2[col]; bias[2][1] = b2[16 + col];

  v16h Bf[6];                                   // mats 3..5 -> frags 6..11
#pragma unroll
  for (int i = 0; i < 6; ++i) Bf[i] = load_b_frag(w16, 6 + i, lane);

  for (int g = blockIdx.x * (blockDim.x >> 5) + wv; g < nG; g += wstride) {
    int node = g * 16 + col;                    // both half-waves stage halves
    _Float16* row = &act[wv][col][0];
    if (node < Nn) {
      const unsigned* ap = agg1 + (size_t)node * 32 + 16 * half;
#pragma unroll
      for (int q = 0; q < 4; ++q) {
        uint4 u = *(const uint4*)(ap + 4 * q);
        row[16 * half + 4 * q + 0] = (_Float16)decf(u.x);
        row[16 * half + 4 * q + 1] = (_Float16)decf(u.y);
        row[16 * half + 4 * q + 2] = (_Float16)decf(u.z);
        row[16 * half + 4 * q + 3] = (_Float16)decf(u.w);
      }
    } else {
      v8h z = {0, 0, 0, 0, 0, 0, 0, 0};
      *(v8h*)(row + 16 * half) = z; *(v8h*)(row + 16 * half + 8) = z;
    }
    __builtin_amdgcn_wave_barrier();

    v16h a = load_a_frag(&act[wv][col][0], half);
    v8f zc = {0.f, 0.f, 0.f, 0.f, 0.f, 0.f, 0.f, 0.f};
    v8f c0 = zc, c1 = zc;
#pragma unroll
    for (int l = 0; l < 3; ++l) {
      c0 = wmma16(a, Bf[2 * l + 0], zc);
      c1 = wmma16(a, Bf[2 * l + 1], zc);
#pragma unroll
      for (int i = 0; i < 8; ++i) { c0[i] += bias[l][0]; c1[i] += bias[l][1]; }
      if (l < 2) {
#pragma unroll
        for (int i = 0; i < 8; ++i) {
          int rr = i + 8 * half;
          act[wv][rr][col]      = (_Float16)eluf(c0[i]);
          act[wv][rr][16 + col] = (_Float16)eluf(c1[i]);
        }
        __builtin_amdgcn_wave_barrier();
        a = load_a_frag(&act[wv][col][0], half);
      }
    }
#pragma unroll
    for (int i = 0; i < 8; ++i) {               // h1 = elu(conv), stored f16
      int nr = g * 16 + i + 8 * half;
      if (nr < Nn) {
        h1[(size_t)nr * 32 + col]      = (_Float16)eluf(c0[i]);
        h1[(size_t)nr * 32 + 16 + col] = (_Float16)eluf(c1[i]);
      }
    }
  }
}

// --------------------------- edge conv, layer 2 ----------------------------
// msg = mlp3_35_35(concat(h1[src], pos[src]-pos[dst])); K padded to 64,
// N padded to 48 (3 N-tiles, 2 K-steps per WMMA chain).
__global__ void __launch_bounds__(256)
gnn_edge2(const _Float16* __restrict__ h1, const float* __restrict__ pos,
          const int* __restrict__ ei, const _Float16* __restrict__ w16,
          const float* __restrict__ b0, const float* __restrict__ b1,
          const float* __restrict__ b2, unsigned* __restrict__ agg2,
          int E, int Nn) {
  __shared__ _Float16 act[8][16][64];
  __shared__ int meta[8][16];
  const int lane = threadIdx.x & 31, wv = threadIdx.x >> 5;
  const int col = lane & 15, half = lane >> 4;
  const int Etot = E + Nn;
  const int nG = (Etot + 15) >> 4;
  const int wstride = (gridDim.x * blockDim.x) >> 5;

  float bias[3][3];
  {
    const float* bs[3] = {b0, b1, b2};
#pragma unroll
    for (int l = 0; l < 3; ++l)
#pragma unroll
      for (int t = 0; t < 3; ++t) {
        int n = t * 16 + col;
        bias[l][t] = (n < 35) ? bs[l][n] : 0.f;
      }
  }

  for (int g = blockIdx.x * (blockDim.x >> 5) + wv; g < nG; g += wstride) {
    int e = g * 16 + col;
    if (lane < 16) {
      bool valid = e < Etot;
      int src = 0, dst = 0;
      if (e < E) { src = ei[e]; dst = ei[E + e]; }
      else if (valid) { src = dst = e - E; }
      meta[wv][col] = valid ? dst : -1;
      _Float16* row = &act[wv][col][0];
      v8h z = {0, 0, 0, 0, 0, 0, 0, 0};
      if (valid) {
        const v8h* hp = (const v8h*)(h1 + (size_t)src * 32);
        *(v8h*)(row + 0)  = hp[0]; *(v8h*)(row + 8)  = hp[1];
        *(v8h*)(row + 16) = hp[2]; *(v8h*)(row + 24) = hp[3];
        *(v8h*)(row + 32) = z; *(v8h*)(row + 40) = z;
        *(v8h*)(row + 48) = z; *(v8h*)(row + 56) = z;
        row[32] = (_Float16)(pos[src * 3 + 0] - pos[dst * 3 + 0]);
        row[33] = (_Float16)(pos[src * 3 + 1] - pos[dst * 3 + 1]);
        row[34] = (_Float16)(pos[src * 3 + 2] - pos[dst * 3 + 2]);
      } else {
#pragma unroll
        for (int q = 0; q < 8; ++q) *(v8h*)(row + 8 * q) = z;
      }
      int pe = e + (wstride << 4);
      if (pe < E) { __builtin_prefetch(ei + pe, 0, 1);
                    __builtin_prefetch(ei + E + pe, 0, 1); }
    }
    __builtin_amdgcn_wave_barrier();

    const _Float16* arow = &act[wv][col][0];
    v16h a0 = load_a_frag(arow, half);
    v16h a1 = load_a_frag(arow + 32, half);
    v8f zc = {0.f, 0.f, 0.f, 0.f, 0.f, 0.f, 0.f, 0.f};
    v8f c[3];
#pragma unroll
    for (int l = 0; l < 3; ++l) {
      int Fb = 12 + 6 * l;                      // mats 6..8 -> frags 12..29
#pragma unroll
      for (int t = 0; t < 3; ++t) {
        v8f acc = wmma16(a0, load_b_frag(w16, Fb + t, lane), zc);
        acc = wmma16(a1, load_b_frag(w16, Fb + 3 + t, lane), acc);
#pragma unroll
        for (int i = 0; i < 8; ++i) acc[i] += bias[l][t];
        c[t] = acc;
      }
      if (l < 2) {
#pragma unroll
        for (int t = 0; t < 3; ++t) {
          int n = t * 16 + col;
#pragma unroll
          for (int i = 0; i < 8; ++i) {
            float hv = (n < 35) ? eluf(c[t][i]) : 0.f;  // keep pad exactly 0
            act[wv][i + 8 * half][n] = (_Float16)hv;
          }
        }
        __builtin_amdgcn_wave_barrier();
        a0 = load_a_frag(arow, half);
        a1 = load_a_frag(arow + 32, half);
      }
    }
#pragma unroll
    for (int i = 0; i < 8; ++i) {
      int d = meta[wv][i + 8 * half];
      if (d >= 0) {
#pragma unroll
        for (int t = 0; t < 3; ++t) {
          int n = t * 16 + col;
          if (n < 35) atomicMax(&agg2[(size_t)d * 48 + n], encf(c[t][i]));
        }
      }
    }
  }
}

// ------------------------ node global MLP, layer 2 -------------------------
// h2 = elu(mlp3_35_35(dec(agg2))); gacc[batch[node]] = max(h2)  (encoded)
__global__ void __launch_bounds__(256)
gnn_node2(const unsigned* __restrict__ agg2, const int* __restrict__ batch,
          const _Float16* __restrict__ w16, const float* __restrict__ b0,
          const float* __restrict__ b1, const float* __restrict__ b2,
          unsigned* __restrict__ gacc, int Nn) {
  __shared__ _Float16 act[8][16][64];
  __shared__ int meta[8][16];
  const int lane = threadIdx.x & 31, wv = threadIdx.x >> 5;
  const int col = lane & 15, half = lane >> 4;
  const int nG = (Nn + 15) >> 4;
  const int wstride = (gridDim.x * blockDim.x) >> 5;

  float bias[3][3];
  {
    const float* bs[3] = {b0, b1, b2};
#pragma unroll
    for (int l = 0; l < 3; ++l)
#pragma unroll
      for (int t = 0; t < 3; ++t) {
        int n = t * 16 + col;
        bias[l][t] = (n < 35) ? bs[l][n] : 0.f;
      }
  }

  for (int g = blockIdx.x * (blockDim.x >> 5) + wv; g < nG; g += wstride) {
    int node = g * 16 + col;
    if (lane < 16) {
      meta[wv][col] = (node < Nn) ? batch[node] : -1;
      _Float16* row = &act[wv][col][0];
      v8h z = {0, 0, 0, 0, 0, 0, 0, 0};
      if (node < Nn) {
        const unsigned* ap = agg2 + (size_t)node * 48;
#pragma unroll
        for (int cq = 0; cq < 48; ++cq)
          row[cq] = (cq < 35) ? (_Float16)decf(ap[cq]) : (_Float16)0.f;
      } else {
#pragma unroll
        for (int q = 0; q < 6; ++q) *(v8h*)(row + 8 * q) = z;
      }
      *(v8h*)(row + 48) = z; *(v8h*)(row + 56) = z;
    }
    __builtin_amdgcn_wave_barrier();

    const _Float16* arow = &act[wv][col][0];
    v16h a0 = load_a_frag(arow, half);
    v16h a1 = load_a_frag(arow + 32, half);
    v8f zc = {0.f, 0.f, 0.f, 0.f, 0.f, 0.f, 0.f, 0.f};
    v8f c[3];
#pragma unroll
    for (int l = 0; l < 3; ++l) {
      int Fb = 30 + 6 * l;                      // mats 9..11 -> frags 30..47
#pragma unroll
      for (int t = 0; t < 3; ++t) {
        v8f acc = wmma16(a0, load_b_frag(w16, Fb + t, lane), zc);
        acc = wmma16(a1, load_b_frag(w16, Fb + 3 + t, lane), acc);
#pragma unroll
        for (int i = 0; i < 8; ++i) acc[i] += bias[l][t];
        c[t] = acc;
      }
      if (l < 2) {
#pragma unroll
        for (int t = 0; t < 3; ++t) {
          int n = t * 16 + col;
#pragma unroll
          for (int i = 0; i < 8; ++i) {
            float hv = (n < 35) ? eluf(c[t][i]) : 0.f;
            act[wv][i + 8 * half][n] = (_Float16)hv;
          }
        }
        __builtin_amdgcn_wave_barrier();
        a0 = load_a_frag(arow, half);
        a1 = load_a_frag(arow + 32, half);
      }
    }
#pragma unroll
    for (int i = 0; i < 8; ++i) {
      int gid = meta[wv][i + 8 * half];
      if (gid >= 0) {
#pragma unroll
        for (int t = 0; t < 3; ++t) {
          int n = t * 16 + col;
          if (n < 35)                           // extra ELU after conv2
            atomicMax(&gacc[(size_t)gid * 48 + n], encf(eluf(c[t][i])));
        }
      }
    }
  }
}

// ------------------------------- final lin ---------------------------------
__global__ void gnn_final(const unsigned* __restrict__ gacc,
                          const float* __restrict__ Wl,
                          const float* __restrict__ bl,
                          float* __restrict__ out, int Gn) {
  int t = blockIdx.x * blockDim.x + threadIdx.x;
  if (t >= Gn * 2) return;
  int g = t >> 1, o = t & 1;
  float s = bl[o];
  for (int k = 0; k < 35; ++k) s += decf(gacc[g * 48 + k]) * Wl[k * 2 + o];
  out[t] = s;
}

// ------------------------------- launcher ----------------------------------
extern "C" void kernel_launch(void* const* d_in, const int* in_sizes, int n_in,
                              void* d_out, int out_size, void* d_ws,
                              size_t ws_size, hipStream_t stream) {
  (void)n_in; (void)ws_size;
  const float* x   = (const float*)d_in[0];
  const float* pos = (const float*)d_in[1];
  const float* W[12];
  const float* Bv[12];
  for (int i = 0; i < 12; ++i) {
    W[i]  = (const float*)d_in[2 + 2 * i];
    Bv[i] = (const float*)d_in[3 + 2 * i];
  }
  const float* Wlin = (const float*)d_in[26];
  const float* blin = (const float*)d_in[27];
  const int* ei     = (const int*)d_in[28];
  const int* batch  = (const int*)d_in[29];

  int Nn = in_sizes[0] / 3;
  int E  = in_sizes[28] / 2;
  int Gn = out_size / 2;

  // workspace carve-up (256B aligned): weights, agg1, agg2, gacc, h1(f16)
  char* base = (char*)d_ws;
  size_t off = 0;
  auto take = [&](size_t bytes) -> char* {
    char* p = base + off;
    off = (off + bytes + 255) & ~(size_t)255;
    return p;
  };
  _Float16* w16  = (_Float16*)take((size_t)48 * 512 * sizeof(_Float16));
  unsigned* agg1 = (unsigned*)take((size_t)Nn * 32 * 4);
  unsigned* agg2 = (unsigned*)take((size_t)Nn * 48 * 4);
  unsigned* gacc = (unsigned*)take((size_t)Gn * 48 * 4);
  _Float16* h1   = (_Float16*)take((size_t)Nn * 32 * sizeof(_Float16));

  int n1 = Nn * 32, n2 = Nn * 48, n3 = Gn * 48;
  gnn_fill<<<(n1 + 255) / 256, 256, 0, stream>>>(agg1, ENC_NEG_INF, n1);
  gnn_fill<<<(n2 + 255) / 256, 256, 0, stream>>>(agg2, ENC_NEG_INF, n2);
  gnn_fill<<<(n3 + 255) / 256, 256, 0, stream>>>(gacc, ENC_NEG_INF, n3);

  PackArgs pa;
  for (int i = 0; i < 12; ++i) pa.w[i] = W[i];
  gnn_pack<<<(48 * 512 + 255) / 256, 256, 0, stream>>>(w16, pa);

  int nGe = (E + Nn + 15) >> 4;
  int eblocks = (nGe + 7) / 8; if (eblocks > 4096) eblocks = 4096;
  int nGn = (Nn + 15) >> 4;
  int nblocks = (nGn + 7) / 8; if (nblocks > 4096) nblocks = 4096;

  gnn_edge1<<<eblocks, 256, 0, stream>>>(x, pos, ei, w16, Bv[0], Bv[1], Bv[2],
                                         agg1, E, Nn);
  gnn_node1<<<nblocks, 256, 0, stream>>>(agg1, w16, Bv[3], Bv[4], Bv[5], h1,
                                         Nn);
  gnn_edge2<<<eblocks, 256, 0, stream>>>(h1, pos, ei, w16, Bv[6], Bv[7], Bv[8],
                                         agg2, E, Nn);
  gnn_node2<<<nblocks, 256, 0, stream>>>(agg2, batch, w16, Bv[9], Bv[10],
                                         Bv[11], gacc, Nn);
  gnn_final<<<(Gn * 2 + 127) / 128, 128, 0, stream>>>(gacc, Wlin, blin,
                                                      (float*)d_out, Gn);
}
